// GumbelSoftmaxVAE_9783935500970
// MI455X (gfx1250) — matrix operand order repaired and checked by reading, not compiled, not run on hardware
//
#include <hip/hip_runtime.h>

typedef __attribute__((ext_vector_type(16))) _Float16 v16h;
typedef __attribute__((ext_vector_type(8)))  float    v8f;

static inline int divup_i(long long a, long long b) { return (int)((a + b - 1) / b); }

// ---------------- utility kernels ----------------

__global__ void fill_f32(float* __restrict__ p, float v, int n) {
    int i = blockIdx.x * blockDim.x + threadIdx.x;
    if (i < n) p[i] = v;
}

__global__ void deg_edges(const int* __restrict__ dst, float* __restrict__ deg, int E) {
    int i = blockIdx.x * blockDim.x + threadIdx.x;
    if (i < E) atomicAdd(&deg[dst[i]], 1.0f);
}

__global__ void rsqrt_ip(float* __restrict__ d, int n) {
    int i = blockIdx.x * blockDim.x + threadIdx.x;
    if (i < n) d[i] = rsqrtf(d[i]);   // deg >= 1 always (self-loop)
}

// convert + transpose weights: Wt[o*K + k] = (f16) W[k*Kout + o]
__global__ void w_to_f16t(const float* __restrict__ W, _Float16* __restrict__ Wt,
                          int K, int Kout) {
    int i = blockIdx.x * blockDim.x + threadIdx.x;
    if (i >= K * Kout) return;
    int k = i / Kout, o = i % Kout;
    Wt[(size_t)o * K + k] = (_Float16)W[i];
}

// ---------------- WMMA GEMM: C[N,Kout] = A[N,K] @ W[K,Kout] (no bias) ----------------
// One wave32 computes a 16x32 output tile (two 16x16 tiles sharing the A fragment).
// K fully unrolled (templated); A loaded as b128 f32 + packed cvt, B pre-converted f16.

template <int K>
__global__ __launch_bounds__(128) void gemm_wmma_f16(
    const float* __restrict__ A, const _Float16* __restrict__ Wt, float* __restrict__ C,
    int N, int Kout) {
    const int lane   = threadIdx.x & 31;
    const int wave   = threadIdx.x >> 5;
    const int tilesP = Kout >> 5;                 // pairs of 16-wide tiles
    const int numTM  = (N + 15) >> 4;
    const int tile   = blockIdx.x * 4 + wave;
    if (tile >= numTM * tilesP) return;           // uniform per-wave: EXEC all-1 at WMMA
    const int tm   = tile / tilesP;
    const int tn   = (tile % tilesP) * 2;
    const int row  = lane & 15;
    const int half = lane >> 4;

    int m_a = tm * 16 + row;
    if (m_a >= N) m_a = N - 1;                    // clamp: OOB rows are never stored
    const int col0 = tn * 16 + row;
    const int col1 = col0 + 16;

    const float*    arow = A  + (size_t)m_a * K;
    const _Float16* b0p  = Wt + (size_t)col0 * K;
    const _Float16* b1p  = Wt + (size_t)col1 * K;

    v8f acc0 = {0.f, 0.f, 0.f, 0.f, 0.f, 0.f, 0.f, 0.f};
    v8f acc1 = {0.f, 0.f, 0.f, 0.f, 0.f, 0.f, 0.f, 0.f};

    #pragma unroll
    for (int k0 = 0; k0 < K; k0 += 32) {
        // A k order used here (contiguous, not AMD's table order): lane half selects K 0-15 (half=0) or 16-31 (half=1), matching B
        const float4 a0 = *reinterpret_cast<const float4*>(arow + k0 + half * 16);
        const float4 a1 = *reinterpret_cast<const float4*>(arow + k0 + half * 16 + 4);
        const float4 a2 = *reinterpret_cast<const float4*>(arow + k0 + half * 16 + 8);
        const float4 a3 = *reinterpret_cast<const float4*>(arow + k0 + half * 16 + 12);
        v16h a;
        a[0]  = (_Float16)a0.x; a[1]  = (_Float16)a0.y; a[2]  = (_Float16)a0.z; a[3]  = (_Float16)a0.w;
        a[4]  = (_Float16)a1.x; a[5]  = (_Float16)a1.y; a[6]  = (_Float16)a1.z; a[7]  = (_Float16)a1.w;
        a[8]  = (_Float16)a2.x; a[9]  = (_Float16)a2.y; a[10] = (_Float16)a2.z; a[11] = (_Float16)a2.w;
        a[12] = (_Float16)a3.x; a[13] = (_Float16)a3.y; a[14] = (_Float16)a3.z; a[15] = (_Float16)a3.w;
        // B k order used here (contiguous, same as A): lanes 0-15 hold K=0..15, lanes 16-31 K=16..31; contiguous in Wt
        const v16h b0 = *reinterpret_cast<const v16h*>(b0p + k0 + half * 16);
        const v16h b1 = *reinterpret_cast<const v16h*>(b1p + k0 + half * 16);
        acc0 = __builtin_amdgcn_wmma_f32_16x16x32_f16(false, a, false, b0, (short)0, acc0, false, false);
        acc1 = __builtin_amdgcn_wmma_f32_16x16x32_f16(false, a, false, b1, (short)0, acc1, false, false);
    }

    // D f32 16x16 layout: VGPR r holds M = r + 8*half, N = lane%16
    #pragma unroll
    for (int r = 0; r < 8; ++r) {
        const int m = tm * 16 + r + 8 * half;
        if (m < N) {
            C[(size_t)m * Kout + col0] = acc0[r];
            C[(size_t)m * Kout + col1] = acc1[r];
        }
    }
}

// ---------------- edge scatter (atomic-add segment sum), float4 feature chunks ----------------

__global__ void scatter_edges_vec4(const float* __restrict__ h, const int* __restrict__ src,
                                   const int* __restrict__ dst, const float* __restrict__ dinv,
                                   float* __restrict__ out, int E, int F) {
    long long tid = (long long)blockIdx.x * blockDim.x + threadIdx.x;
    const int vec = F >> 2;
    const long long total = (long long)E * vec;
    if (tid >= total) return;
    const int e = (int)(tid / vec);
    const int f = (int)(tid % vec) << 2;
    const int s = src[e], d = dst[e];
    const float nrm = dinv[s] * dinv[d];
    const float4 hv = *reinterpret_cast<const float4*>(h + (size_t)s * F + f);
    float* op = out + (size_t)d * F + f;
    atomicAdd(op + 0, nrm * hv.x);
    atomicAdd(op + 1, nrm * hv.y);
    atomicAdd(op + 2, nrm * hv.z);
    atomicAdd(op + 3, nrm * hv.w);
}

__global__ void scatter_edges_f2(const float* __restrict__ h, const int* __restrict__ src,
                                 const int* __restrict__ dst, const float* __restrict__ dinv,
                                 float* __restrict__ out, int E) {
    int e = blockIdx.x * blockDim.x + threadIdx.x;
    if (e >= E) return;
    const int s = src[e], d = dst[e];
    const float nrm = dinv[s] * dinv[d];
    atomicAdd(&out[(size_t)d * 2 + 0], nrm * h[(size_t)s * 2 + 0]);
    atomicAdd(&out[(size_t)d * 2 + 1], nrm * h[(size_t)s * 2 + 1]);
}

// epilogue: out += dinv[n]^2 * t[n,f]  (self-loop message) + bias[f]; optional relu
__global__ void epilogue(float* __restrict__ out, const float* __restrict__ t,
                         const float* __restrict__ dinv, const float* __restrict__ bias,
                         int N, int F, int relu) {
    int i = blockIdx.x * blockDim.x + threadIdx.x;
    if (i >= N * F) return;
    const int n = i / F;
    const int f = i % F;
    const float dn = dinv[n];
    float v = out[i] + dn * dn * t[i] + bias[f];
    if (relu) v = fmaxf(v, 0.f);
    out[i] = v;
}

// ---------------- pointwise tails ----------------

__global__ void reparam(const float* __restrict__ mu, const float* __restrict__ lv,
                        const float* __restrict__ eps, float* __restrict__ z, int n) {
    int i = blockIdx.x * blockDim.x + threadIdx.x;
    if (i < n) z[i] = eps[i] * expf(0.5f * lv[i]) + mu[i];
}

// small dense head: out[n,:Kout] = Z[n,:K] @ W + (bias ? bias : 0)
__global__ void gemm_small(const float* __restrict__ Z, const float* __restrict__ W,
                           const float* __restrict__ bias, float* __restrict__ out,
                           int N, int K, int Kout) {
    int n = blockIdx.x * blockDim.x + threadIdx.x;
    if (n >= N) return;
    for (int o = 0; o < Kout; ++o) {
        float acc = 0.f;
        for (int k = 0; k < K; ++k) acc += Z[(size_t)n * K + k] * W[(size_t)k * Kout + o];
        out[(size_t)n * Kout + o] = acc + (bias ? bias[o] : 0.f);
    }
}

__global__ void gumbel_hard2(const float* __restrict__ lg, const float* __restrict__ u,
                             float* __restrict__ gender, int N) {
    int n = blockIdx.x * blockDim.x + threadIdx.x;
    if (n >= N) return;
    const float g0 = -logf(-logf(u[(size_t)n * 2 + 0] + 1e-20f) + 1e-20f);
    const float g1 = -logf(-logf(u[(size_t)n * 2 + 1] + 1e-20f) + 1e-20f);
    const float a = lg[(size_t)n * 2 + 0] + g0;
    const float b = lg[(size_t)n * 2 + 1] + g1;
    const float m = fmaxf(a, b);
    const float ea = expf(a - m), eb = expf(b - m);
    const float s = ea + eb;
    const float y0 = ea / s, y1 = eb / s;
    const float h0 = (a >= b) ? 1.f : 0.f;   // argmax (first index wins ties)
    const float h1 = 1.f - h0;
    gender[(size_t)n * 2 + 0] = (h0 - y0) + y0;
    gender[(size_t)n * 2 + 1] = (h1 - y1) + y1;
}

// ---------------- launch ----------------

extern "C" void kernel_launch(void* const* d_in, const int* in_sizes, int n_in,
                              void* d_out, int out_size, void* d_ws, size_t ws_size,
                              hipStream_t stream) {
    const float* x   = (const float*)d_in[0];
    const int*   ei  = (const int*)d_in[1];
    const float* eps = (const float*)d_in[2];
    const float* u   = (const float*)d_in[3];
    const float* W1  = (const float*)d_in[4];
    const float* b1  = (const float*)d_in[5];
    const float* Wmu = (const float*)d_in[6];
    const float* bmu = (const float*)d_in[7];
    const float* Wls = (const float*)d_in[8];
    const float* bls = (const float*)d_in[9];
    const float* Wdx = (const float*)d_in[10];
    const float* bdx = (const float*)d_in[11];
    const float* Wg  = (const float*)d_in[12];
    const float* bg  = (const float*)d_in[13];
    const float* Wlc = (const float*)d_in[14];
    const float* blc = (const float*)d_in[15];

    const int F_IN = 128, HID = 128, LAT = 64, OUTF = 128, GEN = 2;
    const int N = in_sizes[0] / F_IN;
    const int E = in_sizes[1] / 2;
    const int* src = ei;
    const int* dst = ei + E;

    // workspace layout: f16 transposed weights first (32KB, 32B-aligned), then floats
    _Float16* Wt_h = (_Float16*)d_ws;                 // up to 128*128 halfs
    float* fbase   = (float*)((char*)d_ws + 128 * 128 * sizeof(_Float16));
    float* dinv = fbase;                              // N
    float* h0   = dinv + N;                           // N*HID
    float* h    = h0 + (size_t)N * HID;               // N*HID
    float* t0   = h + (size_t)N * HID;                // N*OUTF (reused)
    float* t_mu = t0;                                 // N*LAT
    float* t_ls = t0 + (size_t)N * LAT;               // N*LAT
    float* t_dx = t0;                                 // N*OUTF (after reparam)
    float* t_g  = t0 + (size_t)N * OUTF;              // N*GEN
    float* lg   = t_g + (size_t)N * GEN;              // N*GEN

    // output layout (return order)
    float* outp     = (float*)d_out;
    float* o_recon  = outp;                           // N*OUTF
    float* o_gender = o_recon + (size_t)N * OUTF;     // N*GEN
    float* o_label  = o_gender + (size_t)N * GEN;     // N
    float* o_mu     = o_label + N;                    // N*LAT
    float* o_lv     = o_mu + (size_t)N * LAT;         // N*LAT
    float* o_z      = o_lv + (size_t)N * LAT;         // N*LAT

    const int B = 256;
    auto gemm = [&](const float* A, const float* W, float* C, int n, int k, int ko) {
        w_to_f16t<<<divup_i((long long)k * ko, B), B, 0, stream>>>(W, Wt_h, k, ko);
        int tiles = ((n + 15) / 16) * (ko / 32);      // 16x32 tile-pairs per wave
        if (k == 128)
            gemm_wmma_f16<128><<<divup_i(tiles, 4), 128, 0, stream>>>(A, Wt_h, C, n, ko);
        else
            gemm_wmma_f16<64><<<divup_i(tiles, 4), 128, 0, stream>>>(A, Wt_h, C, n, ko);
    };

    // degrees / symmetric norm (deg starts at 1.0 = self-loop)
    fill_f32<<<divup_i(N, B), B, 0, stream>>>(dinv, 1.0f, N);
    deg_edges<<<divup_i(E, B), B, 0, stream>>>(dst, dinv, E);
    rsqrt_ip<<<divup_i(N, B), B, 0, stream>>>(dinv, N);

    // conv1: h = relu(propagate(x@W1) + b1)
    gemm(x, W1, h0, N, F_IN, HID);
    fill_f32<<<divup_i((long long)N * HID, B), B, 0, stream>>>(h, 0.f, N * HID);
    scatter_edges_vec4<<<divup_i((long long)E * (HID / 4), B), B, 0, stream>>>(h0, src, dst, dinv, h, E, HID);
    epilogue<<<divup_i((long long)N * HID, B), B, 0, stream>>>(h, h0, dinv, b1, N, HID, 1);

    // mu / logvar convs
    gemm(h, Wmu, t_mu, N, HID, LAT);
    fill_f32<<<divup_i((long long)N * LAT, B), B, 0, stream>>>(o_mu, 0.f, N * LAT);
    scatter_edges_vec4<<<divup_i((long long)E * (LAT / 4), B), B, 0, stream>>>(t_mu, src, dst, dinv, o_mu, E, LAT);
    epilogue<<<divup_i((long long)N * LAT, B), B, 0, stream>>>(o_mu, t_mu, dinv, bmu, N, LAT, 0);

    gemm(h, Wls, t_ls, N, HID, LAT);
    fill_f32<<<divup_i((long long)N * LAT, B), B, 0, stream>>>(o_lv, 0.f, N * LAT);
    scatter_edges_vec4<<<divup_i((long long)E * (LAT / 4), B), B, 0, stream>>>(t_ls, src, dst, dinv, o_lv, E, LAT);
    epilogue<<<divup_i((long long)N * LAT, B), B, 0, stream>>>(o_lv, t_ls, dinv, bls, N, LAT, 0);

    // z = eps*exp(0.5*logvar) + mu
    reparam<<<divup_i((long long)N * LAT, B), B, 0, stream>>>(o_mu, o_lv, eps, o_z, N * LAT);

    // recon_x conv
    gemm(o_z, Wdx, t_dx, N, LAT, OUTF);
    fill_f32<<<divup_i((long long)N * OUTF, B), B, 0, stream>>>(o_recon, 0.f, N * OUTF);
    scatter_edges_vec4<<<divup_i((long long)E * (OUTF / 4), B), B, 0, stream>>>(t_dx, src, dst, dinv, o_recon, E, OUTF);
    epilogue<<<divup_i((long long)N * OUTF, B), B, 0, stream>>>(o_recon, t_dx, dinv, bdx, N, OUTF, 0);

    // gender head conv + Gumbel hard softmax
    gemm_small<<<divup_i(N, B), B, 0, stream>>>(o_z, Wg, nullptr, t_g, N, LAT, GEN);
    fill_f32<<<divup_i((long long)N * GEN, B), B, 0, stream>>>(lg, 0.f, N * GEN);
    scatter_edges_f2<<<divup_i(E, B), B, 0, stream>>>(t_g, src, dst, dinv, lg, E);
    epilogue<<<divup_i((long long)N * GEN, B), B, 0, stream>>>(lg, t_g, dinv, bg, N, GEN, 0);
    gumbel_hard2<<<divup_i(N, B), B, 0, stream>>>(lg, u, o_gender, N);

    // label head (plain dense)
    gemm_small<<<divup_i(N, B), B, 0, stream>>>(o_z, Wlc, blc, o_label, N, LAT, 1);
}
